// Transformer_34222299415202
// MI455X (gfx1250) — matrix operand order repaired and checked by reading, not compiled, not run on hardware
//
#include <hip/hip_runtime.h>
#include <hip/hip_bf16.h>
#include <math.h>

// ---------------------------------------------------------------------------
// Transformer encoder stack for MI455X (gfx1250, wave32, WMMA).
// B=4, S=1024 (32x32 grid, 3x3 window mask), E=512, H=8, D=64, L=4.
// GEMMs: v_wmma_f32_16x16x32_bf16, 128x128 block tile, 32x64 wave tile,
// double-buffered LDS so global loads overlap the WMMA bursts.
// Attention exploits the 3x3 window (<=9 keys/query) with wave32 shuffles.
// ---------------------------------------------------------------------------

typedef __attribute__((ext_vector_type(16))) __bf16        v16bf;
typedef __attribute__((ext_vector_type(8)))  float         v8f;
typedef __attribute__((ext_vector_type(4)))  float         v4f;
typedef __attribute__((ext_vector_type(4)))  unsigned int  v4u;

#define TK 32
#define LDS_STRIDE (TK + 8)   // 40 bf16 = 80 B row: 16B-aligned frag loads, bank skew

// out[M,N] = act(A[M,K] @ W[N,K]^T + bias[N]) (+ resid[M,N])
// Block: 256 threads (8 waves), tile 128(M) x 128(N). Wave: 32(M) x 64(N).
__global__ __launch_bounds__(256)
void gemm_bias_act_kernel(const float* __restrict__ A,
                          const float* __restrict__ W,
                          const float* __restrict__ bias,
                          const float* __restrict__ resid,
                          float* __restrict__ out,
                          int M, int N, int K, int doGelu)
{
    __shared__ __bf16 As[2][128][LDS_STRIDE];
    __shared__ __bf16 Bs[2][128][LDS_STRIDE];

    const int t    = threadIdx.x;
    const int wave = t >> 5;
    const int lane = t & 31;
    const int wr   = wave >> 1;          // 0..3  -> rows  wr*32..wr*32+31
    const int wc   = wave & 1;           // 0..1  -> cols  wc*64..wc*64+63
    const int bm   = blockIdx.y * 128;
    const int bn   = blockIdx.x * 128;

    // staging: 128x32 f32 per matrix, 256 threads -> 4 v4f each
    const int lrow = t >> 3;             // rows lrow + i*32
    const int lcol = (t & 7) << 2;       // k column within tile

    v4f areg[4], wreg[4];
    auto gload = [&](int kk) {
        #pragma unroll
        for (int i = 0; i < 4; ++i) {
            int row = lrow + i * 32;
            areg[i] = *(const v4f*)&A[(size_t)(bm + row) * K + kk + lcol];
            wreg[i] = *(const v4f*)&W[(size_t)(bn + row) * K + kk + lcol];
        }
    };
    auto sstore = [&](int buf) {
        #pragma unroll
        for (int i = 0; i < 4; ++i) {
            int row = lrow + i * 32;
            As[buf][row][lcol + 0] = (__bf16)areg[i].x;
            As[buf][row][lcol + 1] = (__bf16)areg[i].y;
            As[buf][row][lcol + 2] = (__bf16)areg[i].z;
            As[buf][row][lcol + 3] = (__bf16)areg[i].w;
            Bs[buf][row][lcol + 0] = (__bf16)wreg[i].x;
            Bs[buf][row][lcol + 1] = (__bf16)wreg[i].y;
            Bs[buf][row][lcol + 2] = (__bf16)wreg[i].z;
            Bs[buf][row][lcol + 3] = (__bf16)wreg[i].w;
        }
    };

    const int l16    = lane & 15;
    const int akbase = (lane < 16) ? 0 : 8;    // A frag k-split per lane half
    const int bkbase = (lane < 16) ? 0 : 8;    // B frag k-split per lane half
    const int rbase  = (lane >> 4) << 3;       // C/D: lanes16-31 hold M=r+8

    v8f c[2][4] = {};

    gload(0);
    sstore(0);
    __syncthreads();

    const int nsteps = K / TK;
    for (int s = 0; s < nsteps; ++s) {
        const int buf = s & 1;
        if (s + 1 < nsteps) gload((s + 1) * TK);   // issue next tile's loads early

        // hoist all fragment LDS reads, then burst 8 independent WMMAs
        union frag { v4u q[2]; v16bf v; };
        frag af[2], bfv[4];
        #pragma unroll
        for (int sub = 0; sub < 2; ++sub) {
            int mrow = wr * 32 + sub * 16 + l16;
            af[sub].q[0] = *(const v4u*)&As[buf][mrow][akbase];
            af[sub].q[1] = *(const v4u*)&As[buf][mrow][akbase + 16];
        }
        #pragma unroll
        for (int j = 0; j < 4; ++j) {
            int nrow = wc * 64 + j * 16 + l16;
            bfv[j].q[0] = *(const v4u*)&Bs[buf][nrow][bkbase];
            bfv[j].q[1] = *(const v4u*)&Bs[buf][nrow][bkbase + 16];
        }
        #pragma unroll
        for (int sub = 0; sub < 2; ++sub)
            #pragma unroll
            for (int j = 0; j < 4; ++j)
                c[sub][j] = __builtin_amdgcn_wmma_f32_16x16x32_bf16(
                    false, af[sub].v, false, bfv[j].v, (short)0, c[sub][j],
                    false, false);

        if (s + 1 < nsteps) sstore(buf ^ 1);       // waits on loadcnt AFTER wmma work
        __syncthreads();
    }

    // Epilogue. C/D layout: VGPR r -> M=r (lanes 0-15) / M=r+8 (lanes 16-31); N=lane%16
    #pragma unroll
    for (int sub = 0; sub < 2; ++sub) {
        #pragma unroll
        for (int j = 0; j < 4; ++j) {
            int n = bn + wc * 64 + (j << 4) + l16;
            float bv = bias[n];
            #pragma unroll
            for (int r = 0; r < 8; ++r) {
                int m = bm + wr * 32 + sub * 16 + rbase + r;
                float v = c[sub][j][r] + bv;
                if (doGelu) v = 0.5f * v * (1.0f + erff(v * 0.70710678118654752f));
                if (resid)  v += resid[(size_t)m * N + n];
                out[(size_t)m * N + n] = v;
            }
        }
    }
}

// One wave per row of 512; biased variance, eps=1e-5 (torch LayerNorm).
__global__ __launch_bounds__(128)
void layernorm_kernel(const float* __restrict__ x,
                      const float* __restrict__ g,
                      const float* __restrict__ b,
                      float* __restrict__ out)
{
    const int row  = blockIdx.x * 4 + (threadIdx.x >> 5);
    const int lane = threadIdx.x & 31;
    const float* xr = x + (size_t)row * 512;

    v4f vals[4];
    float s = 0.f, ss = 0.f;
    #pragma unroll
    for (int i = 0; i < 4; ++i) {
        v4f v = *(const v4f*)&xr[(i * 32 + lane) * 4];
        vals[i] = v;
        s  += v.x + v.y + v.z + v.w;
        ss += v.x * v.x + v.y * v.y + v.z * v.z + v.w * v.w;
    }
    #pragma unroll
    for (int off = 16; off > 0; off >>= 1) {
        s  += __shfl_xor(s, off, 32);
        ss += __shfl_xor(ss, off, 32);
    }
    const float mean = s * (1.0f / 512.0f);
    const float var  = ss * (1.0f / 512.0f) - mean * mean;
    const float rstd = rsqrtf(var + 1e-5f);

    float* orow = out + (size_t)row * 512;
    #pragma unroll
    for (int i = 0; i < 4; ++i) {
        int e = (i * 32 + lane) * 4;
        v4f gv = *(const v4f*)&g[e];
        v4f bv = *(const v4f*)&b[e];
        v4f ov;
        ov.x = (vals[i].x - mean) * rstd * gv.x + bv.x;
        ov.y = (vals[i].y - mean) * rstd * gv.y + bv.y;
        ov.z = (vals[i].z - mean) * rstd * gv.z + bv.z;
        ov.w = (vals[i].w - mean) * rstd * gv.w + bv.w;
        *(v4f*)&orow[e] = ov;
    }
}

// Windowed attention: one wave per (b, s, h). 3x3 neighborhood on 32x32 grid.
// qkv: [4096, 1536] rows = b*1024+s, layout [q(512) | k(512) | v(512)], head-major.
__global__ __launch_bounds__(128)
void win_attn_kernel(const float* __restrict__ qkv, float* __restrict__ o)
{
    const int wid  = blockIdx.x * 4 + (threadIdx.x >> 5);  // 0..32767
    const int lane = threadIdx.x & 31;
    const int h    = wid & 7;
    const int s    = (wid >> 3) & 1023;
    const int bb   = wid >> 13;
    const int row  = (bb << 10) + s;
    const int y    = s >> 5;
    const int xg   = s & 31;

    const float* qp = qkv + (size_t)row * 1536 + h * 64;
    const float scale = 0.125f;                  // 1/sqrt(64)
    const float q0 = qp[lane] * scale;
    const float q1 = qp[lane + 32] * scale;

    float sc[9];
    int   krow[9];
    #pragma unroll
    for (int j = 0; j < 9; ++j) {
        int dy = j / 3 - 1, dx = j % 3 - 1;
        int yy = y + dy, xx = xg + dx;
        bool ok = (yy >= 0) && (yy <= 31) && (xx >= 0) && (xx <= 31);
        int yc = min(max(yy, 0), 31);
        int xc = min(max(xx, 0), 31);
        int srow = (bb << 10) + (yc << 5) + xc;
        krow[j] = srow;
        const float* kp = qkv + (size_t)srow * 1536 + 512 + h * 64;
        float p = q0 * kp[lane] + q1 * kp[lane + 32];
        #pragma unroll
        for (int off = 16; off > 0; off >>= 1) p += __shfl_xor(p, off, 32);
        sc[j] = ok ? p : -INFINITY;
    }

    float mx = sc[0];
    #pragma unroll
    for (int j = 1; j < 9; ++j) mx = fmaxf(mx, sc[j]);
    float sum = 0.f;
    #pragma unroll
    for (int j = 0; j < 9; ++j) { sc[j] = expf(sc[j] - mx); sum += sc[j]; }
    const float inv = 1.0f / sum;

    float o0 = 0.f, o1 = 0.f;
    #pragma unroll
    for (int j = 0; j < 9; ++j) {
        const float* vp = qkv + (size_t)krow[j] * 1536 + 1024 + h * 64;
        float p = sc[j] * inv;
        o0 += p * vp[lane];
        o1 += p * vp[lane + 32];
    }
    float* op = o + (size_t)row * 512 + h * 64;
    op[lane]      = o0;
    op[lane + 32] = o1;
}

__global__ void add_kernel(const float* __restrict__ a, const float* __restrict__ b,
                           float* __restrict__ o, int n)
{
    int i = blockIdx.x * blockDim.x + threadIdx.x;
    if (i < n) o[i] = a[i] + b[i];
}

// ---------------------------------------------------------------------------

extern "C" void kernel_launch(void* const* d_in, const int* in_sizes, int n_in,
                              void* d_out, int out_size, void* d_ws, size_t ws_size,
                              hipStream_t stream)
{
    (void)in_sizes; (void)n_in; (void)out_size; (void)ws_size;

    const float* x0     = (const float*)d_in[0];
    // d_in[1] is the mask: window structure is reproduced analytically.
    const float* in_w   = (const float*)d_in[2];
    const float* in_b   = (const float*)d_in[3];
    const float* out_w  = (const float*)d_in[4];
    const float* out_b  = (const float*)d_in[5];
    const float* ln1_g  = (const float*)d_in[6];
    const float* ln1_b  = (const float*)d_in[7];
    const float* ln2_g  = (const float*)d_in[8];
    const float* ln2_b  = (const float*)d_in[9];
    const float* ff1_w  = (const float*)d_in[10];
    const float* ff1_b  = (const float*)d_in[11];
    const float* ff2_w  = (const float*)d_in[12];
    const float* ff2_b  = (const float*)d_in[13];
    const float* mlg    = (const float*)d_in[14];
    const float* mlb    = (const float*)d_in[15];
    const float* mw1    = (const float*)d_in[16];
    const float* mb1    = (const float*)d_in[17];
    const float* mw2    = (const float*)d_in[18];
    const float* mb2    = (const float*)d_in[19];

    float* outp = (float*)d_out;

    const int M = 4096;               // B*S
    const int E = 512;

    float* ws    = (float*)d_ws;
    float* xcur  = ws;                           // [4096, 512]
    float* hbuf  = xcur  + (size_t)M * E;        // [4096, 512]
    float* qkv   = hbuf  + (size_t)M * E;        // [4096, 1536]
    float* attnO = qkv   + (size_t)M * 3 * E;    // [4096, 512]
    float* mid   = attnO + (size_t)M * E;        // [4096, 2048]
    float* x2    = mid   + (size_t)M * 2048;     // [4096, 512]

    hipMemcpyAsync(xcur, x0, (size_t)M * E * sizeof(float),
                   hipMemcpyDeviceToDevice, stream);

    const dim3 blk128(128);
    const dim3 blk256(256);
    const dim3 gLN(M / 4);
    const dim3 gAttn(M * 8 / 4);

    for (int l = 0; l < 4; ++l) {
        const float* iw  = in_w  + (size_t)l * 3 * E * E;
        const float* ib  = in_b  + (size_t)l * 3 * E;
        const float* ow  = out_w + (size_t)l * E * E;
        const float* ob  = out_b + (size_t)l * E;
        const float* l1g = ln1_g + (size_t)l * E;
        const float* l1b = ln1_b + (size_t)l * E;
        const float* l2g = ln2_g + (size_t)l * E;
        const float* l2b = ln2_b + (size_t)l * E;
        const float* f1w = ff1_w + (size_t)l * E * E;
        const float* f1b = ff1_b + (size_t)l * E;
        const float* f2w = ff2_w + (size_t)l * E * E;
        const float* f2b = ff2_b + (size_t)l * E;

        // h = LN1(xcur)
        layernorm_kernel<<<gLN, blk128, 0, stream>>>(xcur, l1g, l1b, hbuf);
        // qkv = h @ in_w^T + in_b            [4096 x 1536]
        gemm_bias_act_kernel<<<dim3(3 * E / 128, M / 128), blk256, 0, stream>>>(
            hbuf, iw, ib, nullptr, qkv, M, 3 * E, E, 0);
        // windowed attention
        win_attn_kernel<<<gAttn, blk128, 0, stream>>>(qkv, attnO);
        // xcur = xcur + attnO @ out_w^T + out_b
        gemm_bias_act_kernel<<<dim3(E / 128, M / 128), blk256, 0, stream>>>(
            attnO, ow, ob, xcur, xcur, M, E, E, 0);
        // h = LN2(xcur)
        layernorm_kernel<<<gLN, blk128, 0, stream>>>(xcur, l2g, l2b, hbuf);
        // mid = gelu(h @ ff1_w^T + ff1_b)    [4096 x 512]
        gemm_bias_act_kernel<<<dim3(E / 128, M / 128), blk256, 0, stream>>>(
            hbuf, f1w, f1b, nullptr, mid, M, E, E, 1);
        // xcur = xcur + mid @ ff2_w^T + ff2_b
        gemm_bias_act_kernel<<<dim3(E / 128, M / 128), blk256, 0, stream>>>(
            mid, f2w, f2b, xcur, xcur, M, E, E, 0);
    }

    // x2 = x0 + xcur  (outer residual around the encoder stack)
    add_kernel<<<dim3(M * E / 256), dim3(256), 0, stream>>>(x0, xcur, x2, M * E);
    // h = LN(x2)
    layernorm_kernel<<<gLN, blk128, 0, stream>>>(x2, mlg, mlb, hbuf);
    // mid = gelu(h @ mlp_w1^T + mlp_b1)      [4096 x 2048]
    gemm_bias_act_kernel<<<dim3(2048 / 128, M / 128), blk256, 0, stream>>>(
        hbuf, mw1, mb1, nullptr, mid, M, 2048, E, 1);
    // out = x2 + mid @ mlp_w2^T + mlp_b2     [4096 x 512], K=2048
    gemm_bias_act_kernel<<<dim3(E / 128, M / 128), blk256, 0, stream>>>(
        mid, mw2, mb2, x2, outp, M, E, 2048, 0);
}
